// Wide_VGG_Small_81973745811644
// MI455X (gfx1250) — matrix operand order repaired and checked by reading, not compiled, not run on hardware
//
#include <hip/hip_runtime.h>
#include <hip/hip_bf16.h>

typedef __attribute__((ext_vector_type(16))) _Float16 v16h;
typedef __attribute__((ext_vector_type(8)))  _Float16 v8h;
typedef __attribute__((ext_vector_type(8)))  float    v8f;

struct __align__(16) H16 { v8h lo, hi; };   // one lane fragment = two adjacent b128

#define BN_EPS 1e-5f
#define MT 128
#define NT 128
#define LDS_STRIDE 40   // f16; 80B row = multiple of 16B so b128 LDS ops stay aligned

// ---------------------------------------------------------------------------
// Layer 1: direct conv (Cin=3) fused BN+sign -> NHWC f16 {-1,0,+1}
// ---------------------------------------------------------------------------
__global__ __launch_bounds__(256) void conv1_bn_sign(
    const float* __restrict__ x, const float* __restrict__ w,
    const float* __restrict__ g, const float* __restrict__ b,
    const float* __restrict__ m, const float* __restrict__ v,
    _Float16* __restrict__ out)
{
    int idx = blockIdx.x * blockDim.x + threadIdx.x;
    if (idx >= 256 * 32 * 32 * 128) return;
    int co = idx & 127;
    int p  = idx >> 7;
    int wq = p & 31;  p >>= 5;
    int h  = p & 31;
    int nimg = p >> 5;

    float acc = 0.f;
    #pragma unroll
    for (int ci = 0; ci < 3; ++ci) {
        #pragma unroll
        for (int r = 0; r < 3; ++r) {
            int ih = h + r - 1;
            if (ih < 0 || ih >= 32) continue;
            #pragma unroll
            for (int s = 0; s < 3; ++s) {
                int iw = wq + s - 1;
                if (iw < 0 || iw >= 32) continue;
                acc += x[((nimg * 3 + ci) * 32 + ih) * 32 + iw] *
                       w[((co * 3 + ci) * 3 + r) * 3 + s];
            }
        }
    }
    float inv = rsqrtf(v[co] + BN_EPS);
    float sc  = g[co] * inv;
    float val = acc * sc + (b[co] - m[co] * sc);
    float sgn = (val > 0.f) ? 1.f : ((val < 0.f) ? -1.f : 0.f);
    out[idx] = (_Float16)sgn;
}

// ---------------------------------------------------------------------------
// Weight repack: fp32 OIHW [co][ci][r][s] -> f16 [co][rs*Cin + ci]
// ---------------------------------------------------------------------------
__global__ void w_repack(const float* __restrict__ src, _Float16* __restrict__ dst,
                         int Cout, int Cin)
{
    int i = blockIdx.x * blockDim.x + threadIdx.x;
    int total = Cout * Cin * 9;
    if (i >= total) return;
    int co  = i / (Cin * 9);
    int rem = i - co * (Cin * 9);
    int ci  = rem / 9;
    int rs  = rem - ci * 9;
    dst[(size_t)co * (Cin * 9) + rs * Cin + ci] = (_Float16)src[i];
}

// ---------------------------------------------------------------------------
// Implicit-GEMM conv3x3, NHWC activations, V_WMMA_F32_16X16X32_F16.
//   Block 256 thr (8 waves): tile 128(M) x 128(N); wave tile 32x64 = 2x4 WMMA.
//   Double-buffered LDS. A tile stored in natural K order so each lane's WMMA
//   fragment is 32 contiguous bytes: lane half h holds K 16h..16h+15 (A and B alike).
//   B tile natural order (per-lane fragment already contiguous).
// ---------------------------------------------------------------------------
__global__ __launch_bounds__(256) void conv_wmma(
    const _Float16* __restrict__ act, const _Float16* __restrict__ wgt,
    _Float16* __restrict__ out,
    const float* __restrict__ bg, const float* __restrict__ bb,
    const float* __restrict__ bm, const float* __restrict__ bv,
    int Cin, int Cout, int H, int W, int fuse)
{
    const int Ktot    = Cin * 9;
    const int HW      = H * W;
    const int cprLog  = __builtin_ctz(Cin >> 5);   // chunks per rs (power of 2)
    const int cprMask = (1 << cprLog) - 1;
    const int nChunks = 9 << cprLog;

    __shared__ __align__(16) _Float16 As[2][MT][LDS_STRIDE];
    __shared__ __align__(16) _Float16 Bs[2][NT][LDS_STRIDE];

    const int t    = threadIdx.x;
    const int lane = t & 31;
    const int wave = t >> 5;
    const int wm   = wave & 3;   // 0..3 : 32-row M sub-tile
    const int wn   = wave >> 2;  // 0..1 : 64-col N sub-tile

    const int m_base = blockIdx.x * MT;
    const int n_base = blockIdx.y * NT;

    // staging: thread -> (row 0..127, 16-element half of the 32-wide K chunk)
    const int srow = t >> 1;
    const int scol = (t & 1) << 4;   // logical K half: 0 or 16
    const int aplo = (t & 1) << 3;   // unused; A low 8 are stored at phys col scol

    // pixel decode for this thread's A staging row (hoisted)
    const int am   = m_base + srow;
    const int a_n  = am / HW;
    const int a_hw = am - a_n * HW;
    const int a_h  = a_hw / W;
    const int a_w  = a_hw - a_h * W;

    const _Float16* b_row = wgt + (size_t)(n_base + srow) * Ktot + scol;

    auto stage = [&](int kc, int buf) {
        int rs = kc >> cprLog;
        int c0 = (kc & cprMask) << 5;
        int r  = rs / 3;
        int s  = rs - r * 3;
        int ih = a_h + r - 1;
        int iw = a_w + s - 1;
        bool valid = (ih >= 0) && (ih < H) && (iw >= 0) && (iw < W);
        v8h alo = (v8h)(_Float16)0.f, ahi = (v8h)(_Float16)0.f;
        if (valid) {
            const _Float16* ap =
                act + (((size_t)a_n * H + ih) * W + iw) * Cin + c0 + scol;
            alo = *(const v8h*)ap;
            ahi = *(const v8h*)(ap + 8);
        }
        // natural store: logical K scol..scol+7 -> phys scol, K scol+8..+15 -> phys scol+8
        *(v8h*)&As[buf][srow][scol]      = alo;
        *(v8h*)&As[buf][srow][scol + 8] = ahi;

        const _Float16* bp = b_row + rs * Cin + c0;
        *(v8h*)&Bs[buf][srow][scol]     = *(const v8h*)bp;
        *(v8h*)&Bs[buf][srow][scol + 8] = *(const v8h*)(bp + 8);
    };

    v8f acc[2][4];
    #pragma unroll
    for (int i = 0; i < 2; ++i)
        #pragma unroll
        for (int j = 0; j < 4; ++j)
            acc[i][j] = (v8f)(0.f);

    // fragment addressing: both A and B (natural order) are 32B contiguous/lane
    const int asel = lane & 15;
    const int fcol = (lane < 16) ? 0 : 16;

    stage(0, 0);
    int buf = 0;
    for (int kc = 0; kc < nChunks; ++kc) {
        __syncthreads();
        if (kc + 1 < nChunks) stage(kc + 1, buf ^ 1);
        if (kc + 2 < nChunks) {
            int kc2 = kc + 2;
            int rs2 = kc2 >> cprLog;
            int c02 = (kc2 & cprMask) << 5;
            __builtin_prefetch(b_row + rs2 * Cin + c02, 0, 0); // global_prefetch_b8
        }

        v16h af[2], bf[4];
        #pragma unroll
        for (int i = 0; i < 2; ++i) {
            const H16* pa = (const H16*)&As[buf][wm * 32 + i * 16 + asel][fcol];
            H16 fa = *pa;
            af[i] = __builtin_shufflevector(fa.lo, fa.hi,
                        0,1,2,3,4,5,6,7,8,9,10,11,12,13,14,15);
        }
        #pragma unroll
        for (int j = 0; j < 4; ++j) {
            const H16* pb = (const H16*)&Bs[buf][wn * 64 + j * 16 + asel][fcol];
            H16 fb = *pb;
            bf[j] = __builtin_shufflevector(fb.lo, fb.hi,
                        0,1,2,3,4,5,6,7,8,9,10,11,12,13,14,15);
        }

        #pragma unroll
        for (int i = 0; i < 2; ++i)
            #pragma unroll
            for (int j = 0; j < 4; ++j)
                acc[i][j] = __builtin_amdgcn_wmma_f32_16x16x32_f16(
                    false, af[i], false, bf[j], (short)0, acc[i][j], false, false);

        buf ^= 1;
    }

    // epilogue: C/D layout -> lanes 0..15: M=e, N=lane; lanes 16..31: M=8+e
    // NHWC output address is m*Cout + ch (no divisions).
    const int nsub  = lane & 15;
    const int mhalf = (lane < 16) ? 0 : 8;
    #pragma unroll
    for (int j = 0; j < 4; ++j) {
        int ch = n_base + wn * 64 + j * 16 + nsub;
        float scale = 1.f, shift = 0.f;
        if (fuse) {
            float inv = rsqrtf(bv[ch] + BN_EPS);
            scale = bg[ch] * inv;
            shift = bb[ch] - bm[ch] * scale;
        }
        #pragma unroll
        for (int i = 0; i < 2; ++i) {
            #pragma unroll
            for (int e = 0; e < 8; ++e) {
                size_t mg = (size_t)(m_base + wm * 32 + i * 16 + mhalf + e);
                float val = acc[i][j][e];
                if (fuse) {
                    val = val * scale + shift;
                    val = (val > 0.f) ? 1.f : ((val < 0.f) ? -1.f : 0.f);
                }
                out[mg * Cout + ch] = (_Float16)val;
            }
        }
    }
}

// ---------------------------------------------------------------------------
// 2x2 maxpool (NHWC raw f16 in), then BN, then:
//   relu_mode=0: sign -> NHWC f16 out
//   relu_mode=1: ReLU -> fp32 out in NCHW-flat order (matches torch .view)
// ---------------------------------------------------------------------------
__global__ __launch_bounds__(256) void pool_bn_post(
    const _Float16* __restrict__ in, _Float16* __restrict__ out_h,
    float* __restrict__ out_f,
    const float* __restrict__ g, const float* __restrict__ b,
    const float* __restrict__ m, const float* __restrict__ v,
    int C, int H, int W, int relu_mode)
{
    int Ho = H >> 1, Wo = W >> 1;
    int total = 256 * C * Ho * Wo;
    int idx = blockIdx.x * blockDim.x + threadIdx.x;
    if (idx >= total) return;
    int c  = idx & (C - 1);          // C is a power of two
    int t2 = idx / C;
    int wo = t2 % Wo;  t2 /= Wo;
    int ho = t2 % Ho;
    int nimg = t2 / Ho;

    const _Float16* p =
        in + (((size_t)nimg * H + ho * 2) * W + wo * 2) * C + c;
    float a0 = (float)p[0];
    float a1 = (float)p[C];
    float a2 = (float)p[(size_t)W * C];
    float a3 = (float)p[(size_t)W * C + C];
    float mx = fmaxf(fmaxf(a0, a1), fmaxf(a2, a3));

    float inv = rsqrtf(v[c] + BN_EPS);
    float sc  = g[c] * inv;
    float val = mx * sc + (b[c] - m[c] * sc);

    if (relu_mode) {
        out_f[(((size_t)nimg * C + c) * Ho + ho) * Wo + wo] = fmaxf(val, 0.f);
    } else {
        float sgn = (val > 0.f) ? 1.f : ((val < 0.f) ? -1.f : 0.f);
        out_h[idx] = (_Float16)sgn;   // idx is NHWC order (c fastest)
    }
}

// ---------------------------------------------------------------------------
// FC head: one wave32 per (batch, class); shuffle reduction
// ---------------------------------------------------------------------------
__global__ __launch_bounds__(320) void fc_kernel(
    const float* __restrict__ fin, const float* __restrict__ fw,
    float* __restrict__ out)
{
    int bidx = blockIdx.x;          // 0..255
    int o    = threadIdx.x >> 5;    // 0..9
    int lane = threadIdx.x & 31;
    const float* xp = fin + (size_t)bidx * 8192;
    const float* wp = fw  + (size_t)o * 8192;
    float s = 0.f;
    for (int k = lane; k < 8192; k += 32)
        s += xp[k] * wp[k];
    #pragma unroll
    for (int off = 16; off > 0; off >>= 1)
        s += __shfl_xor(s, off, 32);
    if (lane == 0) out[bidx * 10 + o] = s;
}

// ---------------------------------------------------------------------------
extern "C" void kernel_launch(void* const* d_in, const int* in_sizes, int n_in,
                              void* d_out, int out_size, void* d_ws, size_t ws_size,
                              hipStream_t stream)
{
    (void)in_sizes; (void)n_in; (void)out_size; (void)ws_size;

    const float* x  = (const float*)d_in[0];
    const float* W1 = (const float*)d_in[1];
    const float* Wf[7]; const float* G[7]; const float* Bb[7];
    const float* Mm[7]; const float* Vv[7];
    for (int i = 1; i <= 6; ++i) {
        Wf[i] = (const float*)d_in[1 + (i - 1) * 5 + 0];
        G[i]  = (const float*)d_in[1 + (i - 1) * 5 + 1];
        Bb[i] = (const float*)d_in[1 + (i - 1) * 5 + 2];
        Mm[i] = (const float*)d_in[1 + (i - 1) * 5 + 3];
        Vv[i] = (const float*)d_in[1 + (i - 1) * 5 + 4];
    }
    const float* fc_w = (const float*)d_in[31];
    float* outp = (float*)d_out;

    // ---- workspace carve-out ----
    char* ws = (char*)d_ws;
    size_t cur = 0;
    auto carve = [&](size_t bytes) -> void* {
        void* p = ws + cur;
        cur = (cur + bytes + 255) & ~(size_t)255;
        return p;
    };
    const int Cins[7]  = {0, 3, 128, 128, 256, 256, 512};
    const int Couts[7] = {0, 128, 128, 256, 256, 512, 512};
    _Float16* Wh[7] = {nullptr};
    for (int i = 2; i <= 6; ++i)
        Wh[i] = (_Float16*)carve((size_t)Couts[i] * Cins[i] * 9 * 2);

    _Float16* ping = (_Float16*)carve((size_t)256 * 128 * 32 * 32 * 2); // 64 MB
    _Float16* pong = (_Float16*)carve((size_t)256 * 128 * 16 * 16 * 2); // 16 MB
    _Float16* raw  = (_Float16*)carve((size_t)256 * 128 * 32 * 32 * 2); // 64 MB
    float*    fcin = (float*)  carve((size_t)256 * 8192 * 4);           //  8 MB

    // ---- weight repack fp32 OIHW -> f16 [co][rs*Cin+ci] ----
    for (int i = 2; i <= 6; ++i) {
        int n = Couts[i] * Cins[i] * 9;
        w_repack<<<(n + 255) / 256, 256, 0, stream>>>(Wf[i], Wh[i], Couts[i], Cins[i]);
    }

    // ---- layer 1: direct conv + BN + sign -> ping (NHWC 256x32x32x128) ----
    conv1_bn_sign<<<(256 * 32 * 32 * 128) / 256, 256, 0, stream>>>(
        x, W1, G[1], Bb[1], Mm[1], Vv[1], ping);

    // ---- layer 2: WMMA conv raw -> pool+BN+sign -> pong (NHWC 16x16x128) ----
    {
        dim3 grid((256 * 32 * 32) / MT, 128 / NT);
        conv_wmma<<<grid, 256, 0, stream>>>(ping, Wh[2], raw, G[2], Bb[2], Mm[2], Vv[2],
                                            128, 128, 32, 32, 0);
        int total = 256 * 128 * 16 * 16;
        pool_bn_post<<<total / 256, 256, 0, stream>>>(raw, pong, nullptr,
                                                      G[2], Bb[2], Mm[2], Vv[2],
                                                      128, 32, 32, 0);
    }

    // ---- layer 3: WMMA conv fused BN+sign -> ping (NHWC 16x16x256) ----
    {
        dim3 grid((256 * 16 * 16) / MT, 256 / NT);
        conv_wmma<<<grid, 256, 0, stream>>>(pong, Wh[3], ping, G[3], Bb[3], Mm[3], Vv[3],
                                            128, 256, 16, 16, 1);
    }

    // ---- layer 4: WMMA conv raw -> pool+BN+sign -> pong (NHWC 8x8x256) ----
    {
        dim3 grid((256 * 16 * 16) / MT, 256 / NT);
        conv_wmma<<<grid, 256, 0, stream>>>(ping, Wh[4], raw, G[4], Bb[4], Mm[4], Vv[4],
                                            256, 256, 16, 16, 0);
        int total = 256 * 256 * 8 * 8;
        pool_bn_post<<<total / 256, 256, 0, stream>>>(raw, pong, nullptr,
                                                      G[4], Bb[4], Mm[4], Vv[4],
                                                      256, 16, 16, 0);
    }

    // ---- layer 5: WMMA conv fused BN+sign -> ping (NHWC 8x8x512) ----
    {
        dim3 grid((256 * 8 * 8) / MT, 512 / NT);
        conv_wmma<<<grid, 256, 0, stream>>>(pong, Wh[5], ping, G[5], Bb[5], Mm[5], Vv[5],
                                            256, 512, 8, 8, 1);
    }

    // ---- layer 6: WMMA conv raw -> pool+BN+ReLU -> fcin (fp32, NCHW flat) ----
    {
        dim3 grid((256 * 8 * 8) / MT, 512 / NT);
        conv_wmma<<<grid, 256, 0, stream>>>(ping, Wh[6], raw, G[6], Bb[6], Mm[6], Vv[6],
                                            512, 512, 8, 8, 0);
        int total = 256 * 512 * 4 * 4;
        pool_bn_post<<<total / 256, 256, 0, stream>>>(raw, nullptr, fcin,
                                                      G[6], Bb[6], Mm[6], Vv[6],
                                                      512, 8, 8, 1);
    }

    // ---- FC head ----
    fc_kernel<<<256, 320, 0, stream>>>(fcin, fc_w, outp);
}
